// LigerFusedLinearJSDLoss_11682311045534
// MI455X (gfx1250) — matrix operand order repaired and checked by reading, not compiled, not run on hardware
//
#include <hip/hip_runtime.h>
#include <hip/hip_bf16.h>

// ---------------------------------------------------------------------------
// Types for the CDNA5 WMMA path
// ---------------------------------------------------------------------------
typedef __attribute__((ext_vector_type(16))) __bf16 bf16x16;
typedef __attribute__((ext_vector_type(8)))  float  f32x8;

union Frag {
    uint4   u[2];   // 32 bytes
    bf16x16 v;      // 16 bf16 = 32 bytes
};

__device__ __forceinline__ unsigned int f32_to_bf16_rne(float f) {
    unsigned int u = __float_as_uint(f);
    unsigned int r = 0x7FFFu + ((u >> 16) & 1u);
    return (u + r) >> 16;
}

// ---------------------------------------------------------------------------
// f32 -> bf16 conversion (pairs, packed dword stores)
// ---------------------------------------------------------------------------
__global__ __launch_bounds__(256)
void cvt_f32_bf16_kernel(const float* __restrict__ src,
                         unsigned short* __restrict__ dst,
                         long long pairs) {
    long long stride = (long long)gridDim.x * blockDim.x;
    for (long long i = (long long)blockIdx.x * blockDim.x + threadIdx.x;
         i < pairs; i += stride) {
        float2 f = ((const float2*)src)[i];
        unsigned int lo = f32_to_bf16_rne(f.x);
        unsigned int hi = f32_to_bf16_rne(f.y);
        ((unsigned int*)dst)[i] = lo | (hi << 16);
    }
}

// ---------------------------------------------------------------------------
// bf16 GEMM via v_wmma_f32_16x16x32_bf16:
//   C[BT x V] = A[BT x K] * W[V x K]^T      (f32 accumulate)
// Workgroup: 256 thr (8 waves), tile 128x128, K-step 32.
// Wave (wm,wn) owns a 64x32 sub-tile = 4x2 tiles of 16x16.
// Staging: double-buffered LDS filled with GLOBAL_LOAD_ASYNC_TO_LDS_B128
// (ASYNCcnt-tracked, no VGPR round-trip); tile t+1 is in flight while tile t
// feeds the WMMAs. Fragments are b128 LDS reads in one contiguous K order
// (k = 16*half + s), the same for A and B.
// ---------------------------------------------------------------------------
__global__ __launch_bounds__(256)
void gemm_bf16_wmma_kernel(const unsigned short* __restrict__ A,
                           const unsigned short* __restrict__ W,
                           float* __restrict__ C,
                           int K, int Vdim) {
    __shared__ uint4 ldsA[2][128 * 4];   // 2 x 8 KB
    __shared__ uint4 ldsB[2][128 * 4];   // 2 x 8 KB

    const int tid  = threadIdx.x;
    const int m0   = blockIdx.y * 128;
    const int n0   = blockIdx.x * 128;
    const int wave = tid >> 5;
    const int lane = tid & 31;
    const int wm   = wave >> 2;       // 0..1 : 64-row slab
    const int wn   = wave & 3;        // 0..3 : 32-col slab
    const int l16  = lane & 15;
    const int half = lane >> 4;       // lane group selects K half

    f32x8 acc[4][2];
#pragma unroll
    for (int i = 0; i < 4; ++i)
#pragma unroll
        for (int j = 0; j < 2; ++j)
            acc[i][j] = {};

    // cooperative staging: each thread owns one 32B chunk of one tile row
    const int srow  = tid >> 1;       // 0..127
    const int shalf = tid & 1;        // 0..1  (K sub-chunk of 16 bf16)
    const unsigned short* gA = A + (size_t)(m0 + srow) * K + shalf * 16;
    const unsigned short* gB = W + (size_t)(n0 + srow) * K + shalf * 16;
    const int sidx = srow * 4 + shalf * 2;

    // LDS byte offsets (low 32 bits of a generic LDS pointer = LDS address)
    unsigned aOff[2], bOff[2];
    aOff[0] = (unsigned)(uintptr_t)&ldsA[0][sidx];
    aOff[1] = (unsigned)(uintptr_t)&ldsA[1][sidx];
    bOff[0] = (unsigned)(uintptr_t)&ldsB[0][sidx];
    bOff[1] = (unsigned)(uintptr_t)&ldsB[1][sidx];

    // Issue 4 async b128 copies (32B of A-tile + 32B of B-tile) into buffer b.
    // INST_OFFSET applies to both LDS and global address (ISA 10.x), so the
    // second 16B chunk reuses the same address VGPRs with offset:16.
    auto issue = [&](int k0, int b) {
        asm volatile(
            "global_load_async_to_lds_b128 %0, %2, off\n\t"
            "global_load_async_to_lds_b128 %0, %2, off offset:16\n\t"
            "global_load_async_to_lds_b128 %1, %3, off\n\t"
            "global_load_async_to_lds_b128 %1, %3, off offset:16"
            :
            : "v"(aOff[b]), "v"(bOff[b]), "v"(gA + k0), "v"(gB + k0)
            : "memory");
    };

    const int NT = K >> 5;
    issue(0, 0);

    for (int t = 0; t < NT; ++t) {
        const int b = t & 1;
        if (t + 1 < NT) {
            issue((t + 1) << 5, b ^ 1);
            // 4 asyncs just issued stay outstanding; previous 4 (tile t) done
            asm volatile("s_wait_asynccnt 4" ::: "memory");
        } else {
            asm volatile("s_wait_asynccnt 0" ::: "memory");
        }
        __syncthreads();   // all threads' tile-t data visible in LDS

        // B fragments: K(32) x N(16); lanes 0-15 hold K=0..15, lanes 16-31
        // hold K=16..31 -> one contiguous 32B chunk per lane.
        Frag bfr[2];
#pragma unroll
        for (int j = 0; j < 2; ++j) {
            int n = wn * 32 + j * 16 + l16;
            bfr[j].u[0] = ldsB[b][n * 4 + half * 2 + 0];
            bfr[j].u[1] = ldsB[b][n * 4 + half * 2 + 1];
        }
        // A fragments: M(16) x K(32); lanes 0-15 hold K=0..15, lanes 16-31
        // hold K=16..31 (same order as B) -> two b128 chunks per lane.
#pragma unroll
        for (int i = 0; i < 4; ++i) {
            int m = wm * 64 + i * 16 + l16;
            Frag af;
            af.u[0] = ldsA[b][m * 4 + half * 2];        // K = half*16 .. half*16+7
            af.u[1] = ldsA[b][m * 4 + half * 2 + 1];    // K = half*16+8 .. half*16+15
#pragma unroll
            for (int j = 0; j < 2; ++j) {
                acc[i][j] = __builtin_amdgcn_wmma_f32_16x16x32_bf16(
                    false, af.v, false, bfr[j].v, (short)0, acc[i][j],
                    false, false);
            }
        }
        __syncthreads();   // protect buffer b before it is re-filled
    }

    // C/D layout: VGPR r -> M=r (lanes 0-15) or M=8+r (lanes 16-31); N=l16
#pragma unroll
    for (int i = 0; i < 4; ++i) {
#pragma unroll
        for (int j = 0; j < 2; ++j) {
            int col  = n0 + wn * 32 + j * 16 + l16;
            int rowb = m0 + wm * 64 + i * 16 + half * 8;
#pragma unroll
            for (int r = 0; r < 8; ++r) {
                C[(size_t)(rowb + r) * Vdim + col] = acc[i][j][r];
            }
        }
    }
}

// ---------------------------------------------------------------------------
// Fused per-row loss: LSE(s), LSE(t), JSD terms, hard CE -- one workgroup per
// row; the row pair (2 x 128 KB) stays L2-resident across the three sweeps,
// so HBM reads each logits matrix only once.
//   acc[0] += sum s_p*(s_lp-log_m)   acc[1] += sum t_p*(t_lp-log_m)
//   acc[2] += valid ? (lse_s - s_logit[label]) : 0
// ---------------------------------------------------------------------------
__global__ __launch_bounds__(256)
void row_loss_kernel(const float* __restrict__ sl, const float* __restrict__ tl,
                     const int* __restrict__ labels, float* __restrict__ acc,
                     int Vdim) {
    __shared__ float r0[256];
    __shared__ float r1[256];
    const int tid = threadIdx.x;
    const int row = blockIdx.x;
    const float* ps = sl + (size_t)row * Vdim;
    const float* pt = tl + (size_t)row * Vdim;
    const int nv4 = Vdim >> 2;
    const float4* ps4 = (const float4*)ps;
    const float4* pt4 = (const float4*)pt;

    // ---- sweep 1: row maxima --------------------------------------------
    float ms = -3.4e38f, mt = -3.4e38f;
    for (int i = tid; i < nv4; i += 256) {
        float4 a = ps4[i], b = pt4[i];
        ms = fmaxf(ms, fmaxf(fmaxf(a.x, a.y), fmaxf(a.z, a.w)));
        mt = fmaxf(mt, fmaxf(fmaxf(b.x, b.y), fmaxf(b.z, b.w)));
    }
    for (int i = (nv4 << 2) + tid; i < Vdim; i += 256) {
        ms = fmaxf(ms, ps[i]);
        mt = fmaxf(mt, pt[i]);
    }
    r0[tid] = ms; r1[tid] = mt;
    __syncthreads();
    for (int s = 128; s > 0; s >>= 1) {
        if (tid < s) {
            r0[tid] = fmaxf(r0[tid], r0[tid + s]);
            r1[tid] = fmaxf(r1[tid], r1[tid + s]);
        }
        __syncthreads();
    }
    ms = r0[0]; mt = r1[0];
    __syncthreads();

    // ---- sweep 2: sum of exp --------------------------------------------
    float ss = 0.f, st = 0.f;
    for (int i = tid; i < nv4; i += 256) {
        float4 a = ps4[i], b = pt4[i];
        ss += expf(a.x - ms) + expf(a.y - ms) + expf(a.z - ms) + expf(a.w - ms);
        st += expf(b.x - mt) + expf(b.y - mt) + expf(b.z - mt) + expf(b.w - mt);
    }
    for (int i = (nv4 << 2) + tid; i < Vdim; i += 256) {
        ss += expf(ps[i] - ms);
        st += expf(pt[i] - mt);
    }
    r0[tid] = ss; r1[tid] = st;
    __syncthreads();
    for (int s = 128; s > 0; s >>= 1) {
        if (tid < s) { r0[tid] += r0[tid + s]; r1[tid] += r1[tid + s]; }
        __syncthreads();
    }
    const float lse_s = ms + logf(r0[0]);
    const float lse_t = mt + logf(r1[0]);
    __syncthreads();

    // ---- sweep 3: JSD terms ---------------------------------------------
    float a_s = 0.f, a_t = 0.f;
    for (int i = tid; i < nv4; i += 256) {
        float4 a = ps4[i], b = pt4[i];
#pragma unroll
        for (int e = 0; e < 4; ++e) {
            float sv = (&a.x)[e], tv = (&b.x)[e];
            float slp = sv - lse_s;
            float tlp = tv - lse_t;
            float sp  = expf(slp);
            float tp  = expf(tlp);
            float lm  = logf(0.5f * (sp + tp));
            a_s = fmaf(sp, slp - lm, a_s);
            a_t = fmaf(tp, tlp - lm, a_t);
        }
    }
    for (int i = (nv4 << 2) + tid; i < Vdim; i += 256) {
        float slp = ps[i] - lse_s;
        float tlp = pt[i] - lse_t;
        float sp  = expf(slp);
        float tp  = expf(tlp);
        float lm  = logf(0.5f * (sp + tp));
        a_s = fmaf(sp, slp - lm, a_s);
        a_t = fmaf(tp, tlp - lm, a_t);
    }
    r0[tid] = a_s; r1[tid] = a_t;
    __syncthreads();
    for (int s = 128; s > 0; s >>= 1) {
        if (tid < s) { r0[tid] += r0[tid + s]; r1[tid] += r1[tid + s]; }
        __syncthreads();
    }
    if (tid == 0) {
        atomicAdd(&acc[0], r0[0]);
        atomicAdd(&acc[1], r1[0]);
        int lbl = labels[row];
        if (lbl != -100)
            atomicAdd(&acc[2], lse_s - ps[lbl]);
    }
}

__global__ void init_kernel(float* acc) {
    if (threadIdx.x < 8) acc[threadIdx.x] = 0.f;
}

__global__ void finalize_kernel(const float* __restrict__ acc,
                                float* __restrict__ out, int BT) {
    float skl  = acc[0];
    float tkl  = acc[1];
    float hard = acc[2];
    float hard_loss = hard / (float)BT;
    float soft_loss = 0.5f * ((skl + tkl) / (float)BT);
    out[0] = 0.5f * hard_loss + 0.5f * soft_loss;
}

// ---------------------------------------------------------------------------
// Launch
// ---------------------------------------------------------------------------
extern "C" void kernel_launch(void* const* d_in, const int* in_sizes, int n_in,
                              void* d_out, int out_size, void* d_ws, size_t ws_size,
                              hipStream_t stream) {
    const float* s_in   = (const float*)d_in[0];
    const float* s_w    = (const float*)d_in[1];
    const float* t_in   = (const float*)d_in[2];
    const float* t_w    = (const float*)d_in[3];
    const int*   labels = (const int*)d_in[4];

    const long long BT = in_sizes[4];
    const long long HS = in_sizes[0] / BT;
    const long long HT = in_sizes[2] / BT;
    const long long V  = (long long)in_sizes[1] / HS;

    char* ws = (char*)d_ws;
    size_t off = 0;
    auto carve = [&](size_t bytes) -> char* {
        char* p = ws + off;
        off = (off + bytes + 255) & ~(size_t)255;
        return p;
    };
    float*          acc    = (float*)carve(8 * sizeof(float));
    unsigned short* s_in_h = (unsigned short*)carve((size_t)BT * HS * 2);
    unsigned short* t_in_h = (unsigned short*)carve((size_t)BT * HT * 2);
    unsigned short* s_w_h  = (unsigned short*)carve((size_t)V * HS * 2);
    unsigned short* t_w_h  = (unsigned short*)carve((size_t)V * HT * 2);
    float*          s_log  = (float*)carve((size_t)BT * V * sizeof(float));
    float*          t_log  = (float*)carve((size_t)BT * V * sizeof(float));
    (void)ws_size; (void)n_in; (void)out_size;

    hipLaunchKernelGGL(init_kernel, dim3(1), dim3(32), 0, stream, acc);

    auto cvt = [&](const float* src, unsigned short* dst, long long n) {
        long long pairs = n / 2;
        long long b = (pairs + 255) / 256;
        if (b > 4096) b = 4096;
        hipLaunchKernelGGL(cvt_f32_bf16_kernel, dim3((unsigned)b), dim3(256), 0,
                           stream, src, dst, pairs);
    };
    cvt(s_in, s_in_h, BT * HS);
    cvt(t_in, t_in_h, BT * HT);
    cvt(s_w,  s_w_h,  V * HS);
    cvt(t_w,  t_w_h,  V * HT);

    dim3 gs((unsigned)(V / 128), (unsigned)(BT / 128));
    hipLaunchKernelGGL(gemm_bf16_wmma_kernel, gs, dim3(256), 0, stream,
                       s_in_h, s_w_h, s_log, (int)HS, (int)V);
    hipLaunchKernelGGL(gemm_bf16_wmma_kernel, gs, dim3(256), 0, stream,
                       t_in_h, t_w_h, t_log, (int)HT, (int)V);

    hipLaunchKernelGGL(row_loss_kernel, dim3((unsigned)BT), dim3(256), 0, stream,
                       s_log, t_log, labels, acc, (int)V);

    hipLaunchKernelGGL(finalize_kernel, dim3(1), dim3(1), 0, stream,
                       acc, (float*)d_out, (int)BT);
}
